// SemanticAdaptiveFilter_773094113755
// MI455X (gfx1250) — matrix operand order repaired and checked by reading, not compiled, not run on hardware
//
#include <hip/hip_runtime.h>
#include <math.h>

#define B_  8
#define T_  1024
#define E_  1024
#define NB_ 10
#define H_  16
#define HD_ 64
#define BT_ (B_*T_)

typedef __bf16 bf16;
typedef __attribute__((ext_vector_type(16))) __bf16 v16bf;
typedef __attribute__((ext_vector_type(8)))  float  v8f;

// ---------------- helpers ----------------

__device__ __forceinline__ unsigned pack2bf(float a, float b) {
  union { bf16 h[2]; unsigned u; } o;
  o.h[0] = (bf16)a; o.h[1] = (bf16)b;
  return o.u;
}

__device__ __forceinline__ v8f wmma_bf16(v16bf a, v16bf b, v8f c) {
  return __builtin_amdgcn_wmma_f32_16x16x32_bf16(false, a, false, b, (short)0, c, false, false);
}

// A-fragment K index for 16-bit 16x32 A matrix (wave32):
// lanes 0-15 half=0: VGPR0..3 -> K 0..7, VGPR4..7 -> K 16..23 ; half=1 adds 8
__device__ __forceinline__ int kofA(int e, int half) {
  return e + half * 8 + (e >= 8 ? 8 : 0);
}

__device__ __forceinline__ float sigmoidf_(float x) { return 1.f / (1.f + expf(-x)); }
__device__ __forceinline__ float geluf_(float x)    { return 0.5f * x * (1.f + erff(x * 0.7071067811865476f)); }

struct Quat { float w, x, y, z; };
__device__ __forceinline__ Quat qmul_(Quat a, Quat b) {
  Quat r;
  r.w = a.w*b.w - a.x*b.x - a.y*b.y - a.z*b.z;
  r.x = a.w*b.x + a.x*b.w + a.y*b.z - a.z*b.y;
  r.y = a.w*b.y - a.x*b.z + a.y*b.w + a.z*b.x;
  r.z = a.w*b.z + a.x*b.y - a.y*b.x + a.z*b.w;
  return r;
}

__device__ __forceinline__ float block_reduce_sum(float v, float* sbuf) {
  const int lane = threadIdx.x & 31;
  const int wv   = threadIdx.x >> 5;
  #pragma unroll
  for (int m = 16; m > 0; m >>= 1) v += __shfl_xor(v, m, 32);
  if (lane == 0) sbuf[wv] = v;
  __syncthreads();
  const int nw = blockDim.x >> 5;
  float r = (threadIdx.x < nw) ? sbuf[threadIdx.x] : 0.f;
  if (wv == 0) {
    #pragma unroll
    for (int m = 4; m > 0; m >>= 1) r += __shfl_xor(r, m, 32);
    if (lane == 0) sbuf[0] = r;
  }
  __syncthreads();
  float out = sbuf[0];
  __syncthreads();
  return out;
}

// ---------------- WMMA GEMM: C = act(A @ W^T + bias) ----------------
// A:[M,K] fp32 (optionally |A-A2|), W:[N,K] fp32, C:[M,N] fp32.
// Block: 256 thr (8 waves, 4x2), tile 128M x 64N, K step 32, bf16 WMMA.
// Staging uses b128 global loads + packed dword LDS stores.
template<int ACT, bool DIFF>
__global__ void gemm_wmma_kernel(const float* __restrict__ A, const float* __restrict__ A2,
                                 const float* __restrict__ W, const float* __restrict__ bias,
                                 float* __restrict__ C, int M, int N, int K) {
  __shared__ bf16 As[128][40];     // 80B row stride: 8B-aligned packed stores, 16B-aligned frag reads
  __shared__ bf16 Bs[32][72];      // 144B row stride: 16B-aligned frag reads
  const int tid  = threadIdx.x;
  const int lane = tid & 31, wv = tid >> 5;
  const int half = lane >> 4, l15 = lane & 15;
  const int wm = wv & 3, wn = wv >> 2;
  const int m0 = blockIdx.y * 128;
  const int n0 = blockIdx.x * 64;

  v8f acc[2][2];
  #pragma unroll
  for (int i = 0; i < 2; i++)
    #pragma unroll
    for (int j = 0; j < 2; j++)
      #pragma unroll
      for (int e = 0; e < 8; e++) acc[i][j][e] = 0.f;

  for (int k0 = 0; k0 < K; k0 += 32) {
    // A tile 128x32 floats: 4 x float4 per thread
    #pragma unroll
    for (int j = 0; j < 4; j++) {
      int idx = tid + 256 * j;
      int r = idx >> 3, c4 = (idx & 7) * 4;
      size_t g = (size_t)(m0 + r) * K + k0 + c4;
      float4 v = *(const float4*)(A + g);
      if (DIFF) {
        float4 u = *(const float4*)(A2 + g);
        v.x = fabsf(v.x - u.x); v.y = fabsf(v.y - u.y);
        v.z = fabsf(v.z - u.z); v.w = fabsf(v.w - u.w);
      }
      unsigned* dst = (unsigned*)&As[r][c4];
      dst[0] = pack2bf(v.x, v.y);
      dst[1] = pack2bf(v.z, v.w);
      if (k0 + 32 < K) __builtin_prefetch(A + g + 32, 0, 1);   // global_prefetch_b8
    }
    // W tile 64x32 -> Bs[k][n] (transposed store)
    #pragma unroll
    for (int j = 0; j < 2; j++) {
      int idx = tid + 256 * j;
      int n = idx >> 3, c4 = (idx & 7) * 4;
      float4 v = {0.f, 0.f, 0.f, 0.f};
      if (n0 + n < N) v = *(const float4*)(W + (size_t)(n0 + n) * K + k0 + c4);
      Bs[c4 + 0][n] = (bf16)v.x;
      Bs[c4 + 1][n] = (bf16)v.y;
      Bs[c4 + 2][n] = (bf16)v.z;
      Bs[c4 + 3][n] = (bf16)v.w;
    }
    __syncthreads();
    v16bf af[2], bfr[2];
    #pragma unroll
    for (int mi = 0; mi < 2; mi++)
      #pragma unroll
      for (int e = 0; e < 16; e++)
        af[mi][e] = As[wm*32 + mi*16 + l15][kofA(e, half)];
    #pragma unroll
    for (int ni = 0; ni < 2; ni++)
      #pragma unroll
      for (int e = 0; e < 16; e++)
        bfr[ni][e] = Bs[kofA(e, half)][wn*32 + ni*16 + l15];
    #pragma unroll
    for (int mi = 0; mi < 2; mi++)
      #pragma unroll
      for (int ni = 0; ni < 2; ni++)
        acc[mi][ni] = wmma_bf16(af[mi], bfr[ni], acc[mi][ni]);
    __syncthreads();
  }

  #pragma unroll
  for (int mi = 0; mi < 2; mi++)
    #pragma unroll
    for (int ni = 0; ni < 2; ni++) {
      int n = n0 + wn*32 + ni*16 + l15;
      if (n < N) {
        #pragma unroll
        for (int e = 0; e < 8; e++) {
          int m = m0 + wm*32 + mi*16 + e + half*8;
          float v = acc[mi][ni][e] + bias[n];
          if (ACT == 1)      v = geluf_(v);
          else if (ACT == 2) v = fmaxf(v, 0.f);
          else if (ACT == 3) v = sigmoidf_(v);
          C[(size_t)m * N + n] = v;
        }
      }
    }
}

// ---------------- LayerNorm + GELU (in place), one row per block ----------------
__global__ void ln_gelu_kernel(float* __restrict__ h, const float* __restrict__ g,
                               const float* __restrict__ bta) {
  __shared__ float sred[32];
  const size_t base = (size_t)blockIdx.x * E_;
  const int tid = threadIdx.x;
  float v[4]; float s = 0.f;
  #pragma unroll
  for (int j = 0; j < 4; j++) { v[j] = h[base + tid + 256*j]; s += v[j]; }
  float mean = block_reduce_sum(s, sred) * (1.f / E_);
  float vs = 0.f;
  #pragma unroll
  for (int j = 0; j < 4; j++) { float d = v[j] - mean; vs += d * d; }
  float var = block_reduce_sum(vs, sred) * (1.f / E_);
  float rstd = rsqrtf(var + 1e-5f);
  #pragma unroll
  for (int j = 0; j < 4; j++) {
    int e = tid + 256*j;
    float y = (v[j] - mean) * rstd * g[e] + bta[e];
    h[base + e] = geluf_(y);
  }
}

// ---------------- quaternion bias correction ----------------
__global__ void bias_correct_kernel(const float* __restrict__ x, const float* __restrict__ sc,
                                    const float* __restrict__ quats, const float* __restrict__ thrp,
                                    float* __restrict__ x2) {
  __shared__ float sq[NB_ * 4];
  __shared__ float ss[NB_];
  const int token = blockIdx.x, tid = threadIdx.x;
  if (tid < NB_ * 4) sq[tid] = quats[tid];
  if (tid < NB_)     ss[tid] = sc[(size_t)token * NB_ + tid];
  __syncthreads();
  const float thr = thrp[0];
  size_t base = (size_t)token * E_ + tid * 4;
  Quat q = { x[base], x[base+1], x[base+2], x[base+3] };
  #pragma unroll
  for (int k = 0; k < NB_; k++) {
    if (ss[k] > thr) {
      Quat bq = { sq[k*4], sq[k*4+1], sq[k*4+2], sq[k*4+3] };
      q = qmul_(bq, q);
    }
  }
  x2[base] = q.w; x2[base+1] = q.x; x2[base+2] = q.y; x2[base+3] = q.z;
}

// ---------------- attention pass 1: per-row max / sum-exp (flash stats) ----------------
// Block 128 thr (4 waves), each wave = 16 queries. grid (B*H, T/64).
__global__ void attn_pass1_kernel(const float* __restrict__ qkv, float* __restrict__ stats) {
  __shared__ bf16 Kt[64][40];
  const int bh = blockIdx.x, qt = blockIdx.y;
  const int b = bh / H_, h = bh % H_;
  const int tid = threadIdx.x, lane = tid & 31, wv = tid >> 5;
  const int half = lane >> 4, l15 = lane & 15;
  const int q0 = qt * 64 + wv * 16;
  const float scale = 0.125f;                       // 1/sqrt(64)
  const size_t rs = 3 * E_;
  const float* qrow = qkv + ((size_t)b * T_ + q0 + l15) * rs + h * HD_;

  v16bf a0, a1;
  #pragma unroll
  for (int e = 0; e < 16; e++) {
    int k = kofA(e, half);
    a0[e] = (bf16)(qrow[k] * scale);
    a1[e] = (bf16)(qrow[32 + k] * scale);
  }

  float mrun[8], lrun[8];
  #pragma unroll
  for (int e = 0; e < 8; e++) { mrun[e] = -3.0e38f; lrun[e] = 0.f; }

  for (int k0 = 0; k0 < T_; k0 += 32) {
    __syncthreads();
    #pragma unroll
    for (int j = 0; j < 4; j++) {                   // K chunk 32x64 via float4
      int idx = tid + 128 * j;
      int key = idx >> 4, c4 = (idx & 15) * 4;
      float4 v = *(const float4*)(qkv + ((size_t)b*T_ + k0 + key) * rs + E_ + h*HD_ + c4);
      Kt[c4 + 0][key] = (bf16)v.x;
      Kt[c4 + 1][key] = (bf16)v.y;
      Kt[c4 + 2][key] = (bf16)v.z;
      Kt[c4 + 3][key] = (bf16)v.w;
    }
    __syncthreads();

    v8f s0, s1;
    #pragma unroll
    for (int e = 0; e < 8; e++) { s0[e] = 0.f; s1[e] = 0.f; }
    v16bf bk;
    #pragma unroll
    for (int e = 0; e < 16; e++) bk[e] = Kt[kofA(e, half)][l15];
    s0 = wmma_bf16(a0, bk, s0);
    #pragma unroll
    for (int e = 0; e < 16; e++) bk[e] = Kt[32 + kofA(e, half)][l15];
    s0 = wmma_bf16(a1, bk, s0);
    #pragma unroll
    for (int e = 0; e < 16; e++) bk[e] = Kt[kofA(e, half)][16 + l15];
    s1 = wmma_bf16(a0, bk, s1);
    #pragma unroll
    for (int e = 0; e < 16; e++) bk[e] = Kt[32 + kofA(e, half)][16 + l15];
    s1 = wmma_bf16(a1, bk, s1);

    #pragma unroll
    for (int e = 0; e < 8; e++) {
      float cm = fmaxf(s0[e], s1[e]);
      cm = fmaxf(cm, __shfl_xor(cm, 1, 32));
      cm = fmaxf(cm, __shfl_xor(cm, 2, 32));
      cm = fmaxf(cm, __shfl_xor(cm, 4, 32));
      cm = fmaxf(cm, __shfl_xor(cm, 8, 32));
      float nm = fmaxf(mrun[e], cm);
      float p = expf(s0[e] - nm) + expf(s1[e] - nm);
      p += __shfl_xor(p, 1, 32);
      p += __shfl_xor(p, 2, 32);
      p += __shfl_xor(p, 4, 32);
      p += __shfl_xor(p, 8, 32);
      lrun[e] = lrun[e] * expf(mrun[e] - nm) + p;
      mrun[e] = nm;
    }
  }
  if (l15 == 0) {
    #pragma unroll
    for (int e = 0; e < 8; e++) {
      int t = q0 + e + half * 8;
      size_t idx = (((size_t)b * H_ + h) * T_ + t) * 2;
      stats[idx] = mrun[e]; stats[idx + 1] = lrun[e];
    }
  }
}

// ---------------- attention pass 2: ctx = P@V, colsum += per-key prob sums ----------------
__global__ void attn_pass2_kernel(const float* __restrict__ qkv, const float* __restrict__ stats,
                                  float* __restrict__ ctx, float* __restrict__ colsum) {
  __shared__ bf16 Kt[64][40];
  __shared__ bf16 Vt[32][72];
  __shared__ bf16 Pt[4][16][40];
  const int bh = blockIdx.x, qt = blockIdx.y;
  const int b = bh / H_, h = bh % H_;
  const int tid = threadIdx.x, lane = tid & 31, wv = tid >> 5;
  const int half = lane >> 4, l15 = lane & 15;
  const int q0 = qt * 64 + wv * 16;
  const float scale = 0.125f;
  const size_t rs = 3 * E_;
  const float* qrow = qkv + ((size_t)b * T_ + q0 + l15) * rs + h * HD_;

  v16bf a0, a1;
  #pragma unroll
  for (int e = 0; e < 16; e++) {
    int k = kofA(e, half);
    a0[e] = (bf16)(qrow[k] * scale);
    a1[e] = (bf16)(qrow[32 + k] * scale);
  }
  float mv[8], linv[8];
  #pragma unroll
  for (int e = 0; e < 8; e++) {
    int t = q0 + e + half * 8;
    size_t idx = (((size_t)b * H_ + h) * T_ + t) * 2;
    mv[e] = stats[idx];
    linv[e] = 1.f / stats[idx + 1];
  }
  v8f cacc[4];
  #pragma unroll
  for (int nf = 0; nf < 4; nf++)
    #pragma unroll
    for (int e = 0; e < 8; e++) cacc[nf][e] = 0.f;

  for (int k0 = 0; k0 < T_; k0 += 32) {
    __syncthreads();
    #pragma unroll
    for (int j = 0; j < 4; j++) {
      int idx = tid + 128 * j;
      int key = idx >> 4, c4 = (idx & 15) * 4;
      size_t row = ((size_t)b*T_ + k0 + key) * rs + h*HD_ + c4;
      float4 v = *(const float4*)(qkv + row + E_);
      Kt[c4 + 0][key] = (bf16)v.x;
      Kt[c4 + 1][key] = (bf16)v.y;
      Kt[c4 + 2][key] = (bf16)v.z;
      Kt[c4 + 3][key] = (bf16)v.w;
      float4 u = *(const float4*)(qkv + row + 2*E_);
      unsigned* dv = (unsigned*)&Vt[key][c4];
      dv[0] = pack2bf(u.x, u.y);
      dv[1] = pack2bf(u.z, u.w);
    }
    __syncthreads();

    v8f s0, s1;
    #pragma unroll
    for (int e = 0; e < 8; e++) { s0[e] = 0.f; s1[e] = 0.f; }
    v16bf bk;
    #pragma unroll
    for (int e = 0; e < 16; e++) bk[e] = Kt[kofA(e, half)][l15];
    s0 = wmma_bf16(a0, bk, s0);
    #pragma unroll
    for (int e = 0; e < 16; e++) bk[e] = Kt[32 + kofA(e, half)][l15];
    s0 = wmma_bf16(a1, bk, s0);
    #pragma unroll
    for (int e = 0; e < 16; e++) bk[e] = Kt[kofA(e, half)][16 + l15];
    s1 = wmma_bf16(a0, bk, s1);
    #pragma unroll
    for (int e = 0; e < 16; e++) bk[e] = Kt[32 + kofA(e, half)][16 + l15];
    s1 = wmma_bf16(a1, bk, s1);

    float p0[8], p1[8], c0 = 0.f, c1 = 0.f;
    #pragma unroll
    for (int e = 0; e < 8; e++) {
      p0[e] = expf(s0[e] - mv[e]) * linv[e];
      p1[e] = expf(s1[e] - mv[e]) * linv[e];
      c0 += p0[e]; c1 += p1[e];
    }
    c0 += __shfl_xor(c0, 16, 32);
    c1 += __shfl_xor(c1, 16, 32);
    if (lane < 16) {
      atomicAdd(&colsum[b * T_ + k0 + lane], c0);
      atomicAdd(&colsum[b * T_ + k0 + 16 + lane], c1);
    }
    #pragma unroll
    for (int e = 0; e < 8; e++) {
      Pt[wv][e + half*8][l15]      = (bf16)p0[e];
      Pt[wv][e + half*8][16 + l15] = (bf16)p1[e];
    }
    __syncthreads();
    v16bf pa;
    #pragma unroll
    for (int e = 0; e < 16; e++) pa[e] = Pt[wv][l15][kofA(e, half)];
    #pragma unroll
    for (int nf = 0; nf < 4; nf++) {
      v16bf bv;
      #pragma unroll
      for (int e = 0; e < 16; e++) bv[e] = Vt[kofA(e, half)][nf*16 + l15];
      cacc[nf] = wmma_bf16(pa, bv, cacc[nf]);
    }
  }
  #pragma unroll
  for (int nf = 0; nf < 4; nf++)
    #pragma unroll
    for (int e = 0; e < 8; e++) {
      int t = q0 + e + half * 8;
      ctx[((size_t)b * T_ + t) * E_ + h * HD_ + nf*16 + l15] = cacc[nf][e];
    }
}

// ---------------- small kernels ----------------
__global__ void zero_kernel(float* __restrict__ p, int n) {
  int i = blockIdx.x * blockDim.x + threadIdx.x;
  if (i < n) p[i] = 0.f;
}

__global__ void qmean_kernel(const float* __restrict__ x2, float* __restrict__ qm) {
  __shared__ float sred[32];
  const int token = blockIdx.x, tid = threadIdx.x;
  size_t base = (size_t)token * E_ + tid * 4;
  float w = x2[base], x = x2[base+1], y = x2[base+2], z = x2[base+3];
  float inv = 1.f / (sqrtf(w*w + x*x + y*y + z*z) + 1e-8f);
  w *= inv; x *= inv; y *= inv; z *= inv;
  float rw = block_reduce_sum(w, sred);
  float rx = block_reduce_sum(x, sred);
  float ry = block_reduce_sum(y, sred);
  float rz = block_reduce_sum(z, sred);
  if (tid == 0) {
    qm[token*4+0] = rw * (1.f/256.f);
    qm[token*4+1] = rx * (1.f/256.f);
    qm[token*4+2] = ry * (1.f/256.f);
    qm[token*4+3] = rz * (1.f/256.f);
  }
}

__global__ void opp_kernel(const float* __restrict__ qm, float* __restrict__ opp) {
  int idx = blockIdx.x * blockDim.x + threadIdx.x;
  if (idx >= BT_) return;
  int t = idx % T_;
  float o = 0.f;
  if (t > 0) {
    float d = 0.f;
    #pragma unroll
    for (int c = 0; c < 4; c++) d += qm[idx*4 + c] * qm[(idx-1)*4 + c];
    o = fminf(fmaxf(-d, 0.f), 1.f);
  }
  opp[idx] = o;
}

__global__ void anomaly_kernel(const float* __restrict__ x2, float* __restrict__ anom) {
  __shared__ float sred[32];
  const int token = blockIdx.x, tid = threadIdx.x;
  const int t = token % T_;
  const int b = token / T_;
  const int st = (t - 2 > 0) ? t - 2 : 0;
  const int en = (t + 3 < T_) ? t + 3 : T_;
  const float n = (float)(en - st);
  const float* xb = x2 + (size_t)b * T_ * E_;
  float accum = 0.f;
  #pragma unroll
  for (int j = 0; j < 4; j++) {
    int e = tid + 256 * j;
    float s = 0.f, ss = 0.f;
    for (int r = st; r < en; r++) {
      float v = xb[(size_t)r * E_ + e];
      s += v; ss += v * v;
    }
    float mean = s / n;
    float sd = sqrtf(fmaxf((ss - s * s / n) / (n - 1.f), 0.f)) + 1e-8f;
    accum += fabsf(xb[(size_t)t * E_ + e] - mean) / sd;
  }
  float tot = block_reduce_sum(accum, sred);
  if (tid == 0) anom[token] = tot * (1.f / E_);
}

__global__ void wa_kernel(const float* __restrict__ colsum, const float* __restrict__ opp,
                          float* __restrict__ wa) {
  __shared__ float sred[32];
  const int b = blockIdx.x, tid = threadIdx.x;
  float s = 0.f;
  for (int t = tid; t < T_; t += 256)
    s += colsum[b * T_ + t] * opp[b * T_ + t];
  float tot = block_reduce_sum(s, sred);
  if (tid == 0) wa[b] = tot * (1.f / ((float)H_ * (float)T_));
}

__global__ void cscore_kernel(const float* __restrict__ opp, const float* __restrict__ anom,
                              const float* __restrict__ base, const float* __restrict__ wa,
                              float* __restrict__ csc) {
  __shared__ float sred[32];
  const int b = blockIdx.x, tid = threadIdx.x;
  const float wab = wa[b];
  float so = 0, qo = 0, sa = 0, qa = 0, sn = 0, qn = 0;
  for (int t = tid; t < T_; t += 256) {
    float on = sigmoidf_(opp[b*T_ + t] * 2.f);
    float an = sigmoidf_(((t < T_ - 1) ? wab : 0.f) * 2.f);
    float nn = sigmoidf_(anom[b*T_ + t] - 1.f);
    so += on; qo += on * on; sa += an; qa += an * an; sn += nn; qn += nn * nn;
  }
  so = block_reduce_sum(so, sred); qo = block_reduce_sum(qo, sred);
  sa = block_reduce_sum(sa, sred); qa = block_reduce_sum(qa, sred);
  sn = block_reduce_sum(sn, sred); qn = block_reduce_sum(qn, sred);
  const float Tf = (float)T_;
  float stdo = sqrtf(fmaxf((qo - so*so/Tf) / (Tf - 1.f), 0.f)) + 0.1f;
  float stda = sqrtf(fmaxf((qa - sa*sa/Tf) / (Tf - 1.f), 0.f)) + 0.1f;
  float stdn = sqrtf(fmaxf((qn - sn*sn/Tf) / (Tf - 1.f), 0.f)) + 0.1f;
  for (int t = tid; t < T_; t += 256) {
    float on = sigmoidf_(opp[b*T_ + t] * 2.f);
    float an = sigmoidf_(((t < T_ - 1) ? wab : 0.f) * 2.f);
    float nn = sigmoidf_(anom[b*T_ + t] - 1.f);
    float comb = (on * stdo + an * stda + nn * stdn) * (1.f / 3.f);
    csc[b*T_ + t] = sigmoidf_((base[b*T_ + t] + comb - 0.5f) * 2.f);
  }
}

__global__ void final_kernel(const float* __restrict__ x2, const float* __restrict__ csc,
                             const float* __restrict__ pf, float* __restrict__ out) {
  const int token = blockIdx.x, tid = threadIdx.x;
  const float cs = csc[token];
  float inten = fminf(fmaxf((cs - 0.3f) / 0.7f, 0.f), 1.f);
  float strength = 0.5f * inten;
  float a0 = pf[0] * strength, a1 = pf[1] * strength, a2 = pf[2] * strength;
  float ch = cosf(a0 * 0.5f), sh = sinf(a0 * 0.5f);
  Quat aq = { ch, sh * cosf(a1), sh * sinf(a1) * cosf(a2), sh * sinf(a1) * sinf(a2) };
  float pres = 1.f - strength * 0.5f;
  bool apply = inten > 0.1f;
  size_t base = (size_t)token * E_ + tid * 4;
  Quat q = { x2[base], x2[base+1], x2[base+2], x2[base+3] };
  Quat r = qmul_(aq, q);
  Quat o;
  if (apply) {
    o.w = pres*q.w + (1.f - pres)*r.w;
    o.x = pres*q.x + (1.f - pres)*r.x;
    o.y = pres*q.y + (1.f - pres)*r.y;
    o.z = pres*q.z + (1.f - pres)*r.z;
  } else o = q;
  out[base] = o.w; out[base+1] = o.x; out[base+2] = o.y; out[base+3] = o.z;
}

// ---------------- orchestration ----------------
extern "C" void kernel_launch(void* const* d_in, const int* in_sizes, int n_in,
                              void* d_out, int out_size, void* d_ws, size_t ws_size,
                              hipStream_t stream) {
  (void)in_sizes; (void)n_in; (void)out_size; (void)ws_size;
  const float* x        = (const float*)d_in[0];
  const float* w1       = (const float*)d_in[1];
  const float* b1       = (const float*)d_in[2];
  const float* ln_g     = (const float*)d_in[3];
  const float* ln_b     = (const float*)d_in[4];
  const float* w2       = (const float*)d_in[5];
  const float* b2       = (const float*)d_in[6];
  const float* w3       = (const float*)d_in[7];
  const float* b3       = (const float*)d_in[8];
  const float* bquats   = (const float*)d_in[9];
  const float* bthresh  = (const float*)d_in[10];
  const float* w_in     = (const float*)d_in[11];
  const float* b_in     = (const float*)d_in[12];
  const float* w_out    = (const float*)d_in[13];
  const float* b_out    = (const float*)d_in[14];
  const float* ws1      = (const float*)d_in[15];
  const float* bs1      = (const float*)d_in[16];
  const float* ws2      = (const float*)d_in[17];
  const float* bs2      = (const float*)d_in[18];
  const float* pfilter  = (const float*)d_in[19];
  float* out = (float*)d_out;

  float* f = (float*)d_ws;
  size_t o = 0;
  float* bufA   = f + o; o += (size_t)BT_ * E_;        // h / ctx
  float* bufB   = f + o; o += (size_t)BT_ * (E_/2);    // h2 / sc1
  float* bscore = f + o; o += (size_t)BT_ * NB_;
  float* x2     = f + o; o += (size_t)BT_ * E_;
  float* qkvb   = f + o; o += (size_t)BT_ * 3 * E_;    // qkv, later attn_out
  float* stats  = f + o; o += (size_t)B_ * H_ * T_ * 2;
  float* colsum = f + o; o += BT_;
  float* qmean  = f + o; o += (size_t)BT_ * 4;
  float* opp    = f + o; o += BT_;
  float* anom   = f + o; o += BT_;
  float* baseb  = f + o; o += BT_;
  float* wab    = f + o; o += B_;
  float* csc    = f + o; o += BT_;

  dim3 blk256(256), blk128(128);

  // 1) h = x @ w1^T + b1
  gemm_wmma_kernel<0, false><<<dim3(E_/64, BT_/128), blk256, 0, stream>>>(x, nullptr, w1, b1, bufA, BT_, E_, E_);
  // 2) LayerNorm + GELU in place
  ln_gelu_kernel<<<BT_, blk256, 0, stream>>>(bufA, ln_g, ln_b);
  // 3) h2 = gelu(h @ w2^T + b2)
  gemm_wmma_kernel<1, false><<<dim3((E_/2)/64, BT_/128), blk256, 0, stream>>>(bufA, nullptr, w2, b2, bufB, BT_, E_/2, E_);
  // 4) bias_scores = sigmoid(h2 @ w3^T + b3)
  gemm_wmma_kernel<3, false><<<dim3(1, BT_/128), blk256, 0, stream>>>(bufB, nullptr, w3, b3, bscore, BT_, NB_, E_/2);
  // 5) quaternion bias correction
  bias_correct_kernel<<<BT_, blk256, 0, stream>>>(x, bscore, bquats, bthresh, x2);
  // 6) qkv = x2 @ w_in^T + b_in
  gemm_wmma_kernel<0, false><<<dim3((3*E_)/64, BT_/128), blk256, 0, stream>>>(x2, nullptr, w_in, b_in, qkvb, BT_, 3*E_, E_);
  // 7) flash stats
  attn_pass1_kernel<<<dim3(B_*H_, T_/64), blk128, 0, stream>>>(qkvb, stats);
  // 8) zero colsum
  zero_kernel<<<(BT_ + 255)/256, blk256, 0, stream>>>(colsum, BT_);
  // 9) ctx + column prob sums
  attn_pass2_kernel<<<dim3(B_*H_, T_/64), blk128, 0, stream>>>(qkvb, stats, bufA, colsum);
  // 10) attn_out = ctx @ w_out^T + b_out (reuse qkv buffer)
  float* attn_out = qkvb;
  gemm_wmma_kernel<0, false><<<dim3(E_/64, BT_/128), blk256, 0, stream>>>(bufA, nullptr, w_out, b_out, attn_out, BT_, E_, E_);
  // 11) sc1 = relu(|attn_out - x2| @ ws1^T + bs1)
  gemm_wmma_kernel<2, true><<<dim3((E_/2)/64, BT_/128), blk256, 0, stream>>>(attn_out, x2, ws1, bs1, bufB, BT_, E_/2, E_);
  // 12) base = sigmoid(sc1 @ ws2^T + bs2)
  gemm_wmma_kernel<3, false><<<dim3(1, BT_/128), blk256, 0, stream>>>(bufB, nullptr, ws2, bs2, baseb, BT_, 1, E_/2);
  // 13) normalized quaternion means
  qmean_kernel<<<BT_, blk256, 0, stream>>>(x2, qmean);
  // 14) opposition
  opp_kernel<<<(BT_ + 255)/256, blk256, 0, stream>>>(qmean, opp);
  // 15) windowed anomalies
  anomaly_kernel<<<BT_, blk256, 0, stream>>>(x2, anom);
  // 16) wa = sum(am * opposition)
  wa_kernel<<<B_, blk256, 0, stream>>>(colsum, opp, wab);
  // 17) contradiction scores
  cscore_kernel<<<B_, blk256, 0, stream>>>(opp, anom, baseb, wab, csc);
  // 18) phase-filter output
  final_kernel<<<BT_, blk256, 0, stream>>>(x2, csc, pfilter, out);
}
